// MaskTransformerNet_7335804142071
// MI455X (gfx1250) — compile-verified, matrix operand order checked by reading, not run on hardware
//
#include <hip/hip_runtime.h>
#include <hip/hip_bf16.h>
#include <stdint.h>

typedef _Float16 h16;
typedef __attribute__((ext_vector_type(16))) _Float16 v16h;
typedef __attribute__((ext_vector_type(8)))  _Float16 v8h;
typedef __attribute__((ext_vector_type(8)))  float    v8f;

// ---------------------------------------------------------------------------
// Weight prep: W[M,K] f32 row-major (optionally scaled per-row by g[m]*invBN)
// -> Wp[Mp,Kp] f16 row-major, zero padded.
// ---------------------------------------------------------------------------
__global__ void k_prep_weight(const float* __restrict__ W, const float* __restrict__ g,
                              h16* __restrict__ Wp, int M, int K, int Mp, int Kp,
                              float inv) {
  int i = blockIdx.x * blockDim.x + threadIdx.x;
  if (i >= Mp * Kp) return;
  int m = i / Kp, k = i % Kp;
  float v = 0.0f;
  if (m < M && k < K) {
    v = W[(size_t)m * K + k];
    if (g) v *= g[m] * inv;
  }
  Wp[i] = (h16)v;
}

// Column-major (C,N) f32 -> row f16 [N, Cp]
__global__ void k_cm_to_rows(const float* __restrict__ x, int C, int N,
                             h16* __restrict__ rows, int Cp) {
  int i = blockIdx.x * blockDim.x + threadIdx.x;
  if (i >= N * Cp) return;
  int n = i / Cp, c = i % Cp;
  rows[i] = (c < C) ? (h16)x[(size_t)c * N + n] : (h16)0.0f;
}

// xyz gather: dst(3,S) = src(3,N)[:, perm]
__global__ void k_gather_xyz(const float* __restrict__ src, int N,
                             const int* __restrict__ perm,
                             float* __restrict__ dst, int S) {
  int i = blockIdx.x * blockDim.x + threadIdx.x;
  if (i >= 3 * S) return;
  int d = i / S, s = i % S;
  dst[i] = src[(size_t)d * N + perm[s]];
}

// ---------------------------------------------------------------------------
// KNN: one thread per query; register-resident sorted top-k (smallest d2).
// ---------------------------------------------------------------------------
template <int KNB>
__global__ void k_knn(const float* __restrict__ sup, int Ns,
                      const float* __restrict__ qry, int Nq,
                      int* __restrict__ idx) {
  int q = blockIdx.x * blockDim.x + threadIdx.x;
  if (q >= Nq) return;
  float qx = qry[q], qy = qry[Nq + q], qz = qry[2 * Nq + q];
  float bd[KNB];
  int   bi[KNB];
#pragma unroll
  for (int i = 0; i < KNB; ++i) { bd[i] = 3.4e38f; bi[i] = 0; }
  for (int s = 0; s < Ns; ++s) {
    float dx = sup[s] - qx;
    float dy = sup[Ns + s] - qy;
    float dz = sup[2 * Ns + s] - qz;
    float d = dx * dx + dy * dy + dz * dz;
    if (d < bd[KNB - 1]) {
      bd[KNB - 1] = d; bi[KNB - 1] = s;
#pragma unroll
      for (int j = KNB - 1; j > 0; --j) {
        if (bd[j] < bd[j - 1]) {
          float td = bd[j]; bd[j] = bd[j - 1]; bd[j - 1] = td;
          int   ti = bi[j]; bi[j] = bi[j - 1]; bi[j - 1] = ti;
        }
      }
    }
  }
#pragma unroll
  for (int i = 0; i < KNB; ++i) idx[(size_t)q * KNB + i] = bi[i];
}

// ---------------------------------------------------------------------------
// Grouping: build Xg[L=S*K, Kpad] f16 rows: [feats(C), rel_xyz(3), pad]
// ---------------------------------------------------------------------------
__global__ void k_group(const h16* __restrict__ featrows, int Cp, int C,
                        const float* __restrict__ xyz, int N,
                        const float* __restrict__ nxyz, int S,
                        const int* __restrict__ idx, int K,
                        h16* __restrict__ Xg, int Kpad) {
  int j = blockIdx.x * blockDim.x + threadIdx.x;
  if (j >= S * K) return;
  int s = j / K;
  int n = idx[j];
  h16* out = Xg + (size_t)j * Kpad;
  const h16* fr = featrows + (size_t)n * Cp;
  for (int c = 0; c < C; ++c) out[c] = fr[c];
  for (int d = 0; d < 3; ++d)
    out[C + d] = (h16)(xyz[(size_t)d * N + n] - nxyz[(size_t)d * S + s]);
  for (int c = C + 3; c < Kpad; ++c) out[c] = (h16)0.0f;
}

// ---------------------------------------------------------------------------
// WMMA GEMM, NT output tiles per wave (16 x 16*NT strip, shared A fragment):
//   Y[j,m] = act( sum_k Xg[j,k]*Wp[m,k] + bias[m] )
// grid = (L/(16*NT), Mp/16), block = 32 (one wave). Kp % 32 == 0.
// Fragment layout per CDNA5 ISA 7.12.2: lanes 0-15 hold row lm with
// K {k0..k0+7, k0+16..k0+23}; lanes 16-31 the +8 K halves.
// ---------------------------------------------------------------------------
template <int NT>
__global__ void k_gemm(const h16* __restrict__ Wp, const h16* __restrict__ Xg,
                       const float* __restrict__ bias,
                       h16* __restrict__ Yh, float* __restrict__ Yf,
                       int Mp, int Kp, int relu) {
  const int lane = threadIdx.x;
  const int hi = lane >> 4;
  const int lm = lane & 15;
  const h16* arow = Wp + (size_t)(blockIdx.y * 16 + lm) * Kp + hi * 8;
  const h16* brow[NT];
#pragma unroll
  for (int t = 0; t < NT; ++t)
    brow[t] = Xg + (size_t)(blockIdx.x * (16 * NT) + t * 16 + lm) * Kp + hi * 8;

  v8f zero = {};
  v8f acc[NT];
#pragma unroll
  for (int t = 0; t < NT; ++t) acc[t] = zero;

  for (int k0 = 0; k0 < Kp; k0 += 32) {
    v8h a0 = *(const v8h*)(arow + k0);
    v8h a1 = *(const v8h*)(arow + k0 + 16);
    v16h A = __builtin_shufflevector(a0, a1, 0, 1, 2, 3, 4, 5, 6, 7,
                                     8, 9, 10, 11, 12, 13, 14, 15);
#pragma unroll
    for (int t = 0; t < NT; ++t) {
      v8h b0 = *(const v8h*)(brow[t] + k0);
      v8h b1 = *(const v8h*)(brow[t] + k0 + 16);
      __builtin_prefetch(brow[t] + k0 + 64, 0, 1);  // K tile two ahead
      v16h B = __builtin_shufflevector(b0, b1, 0, 1, 2, 3, 4, 5, 6, 7,
                                       8, 9, 10, 11, 12, 13, 14, 15);
      acc[t] = __builtin_amdgcn_wmma_f32_16x16x32_f16(false, A, false, B,
                                                      (short)0, acc[t],
                                                      false, false);
    }
  }

  const int m0 = blockIdx.y * 16 + hi * 8;
  float bb[8];
#pragma unroll
  for (int r = 0; r < 8; ++r) bb[r] = bias ? bias[m0 + r] : 0.0f;

#pragma unroll
  for (int t = 0; t < NT; ++t) {
    const size_t j = (size_t)blockIdx.x * (16 * NT) + t * 16 + lm;
    float o[8];
#pragma unroll
    for (int r = 0; r < 8; ++r) {
      float v = acc[t][r] + bb[r];
      if (relu) v = fmaxf(v, 0.0f);
      o[r] = v;
    }
    if (Yh) {
      v8h hh;
#pragma unroll
      for (int r = 0; r < 8; ++r) hh[r] = (h16)o[r];
      *(v8h*)(Yh + j * Mp + m0) = hh;
    }
    if (Yf) {
#pragma unroll
      for (int r = 0; r < 8; ++r) Yf[j * Mp + m0 + r] = o[r];
    }
  }
}

// max over K neighbors: Y[S*K, Mp] -> out[S, Mp]
__global__ void k_maxk(const h16* __restrict__ Y, h16* __restrict__ out,
                       int S, int K, int Mp) {
  int i = blockIdx.x * blockDim.x + threadIdx.x;
  if (i >= S * Mp) return;
  int s = i / Mp, c = i % Mp;
  float mx = -3.4e38f;
  for (int t = 0; t < K; ++t)
    mx = fmaxf(mx, (float)Y[((size_t)s * K + t) * Mp + c]);
  out[(size_t)s * Mp + c] = (h16)mx;
}

// ---------------------------------------------------------------------------
// Feature-prop: inverse-distance interp of coarse feats + concat fine feats.
// ---------------------------------------------------------------------------
__global__ void k_interp_concat(const float* __restrict__ xyzf, int Nq,
                                const float* __restrict__ xyzc, int Nc,
                                const int* __restrict__ idx, int kk,
                                const h16* __restrict__ fc, int Cpc, int Cc,
                                const h16* __restrict__ ff, int Cpf, int Cf,
                                h16* __restrict__ Xcat, int Kpad) {
  int j = blockIdx.x * blockDim.x + threadIdx.x;
  if (j >= Nq) return;
  float w[3]; int id[3];
  float sw = 0.0f;
  for (int t = 0; t < kk; ++t) {
    int n = idx[(size_t)j * kk + t];
    id[t] = n;
    float dx = xyzf[j] - xyzc[n];
    float dy = xyzf[Nq + j] - xyzc[Nc + n];
    float dz = xyzf[2 * Nq + j] - xyzc[2 * Nc + n];
    float d = sqrtf(dx * dx + dy * dy + dz * dz + 1e-16f);
    w[t] = 1.0f / d;
    sw += w[t];
  }
  for (int t = 0; t < kk; ++t) w[t] /= sw;
  h16* out = Xcat + (size_t)j * Kpad;
  for (int c = 0; c < Cc; ++c) {
    float s = 0.0f;
    for (int t = 0; t < kk; ++t) s += w[t] * (float)fc[(size_t)id[t] * Cpc + c];
    out[c] = (h16)s;
  }
  const h16* fr = ff + (size_t)j * Cpf;
  for (int c = 0; c < Cf; ++c) out[Cc + c] = fr[c];
  for (int c = Cc + Cf; c < Kpad; ++c) out[c] = (h16)0.0f;
}

// ---------------------------------------------------------------------------
// Decoder small kernels (D=128, wave32)
// ---------------------------------------------------------------------------
__global__ void k_init_q(const float* __restrict__ qe, float* __restrict__ q32,
                         int NQ, int D, int R) {
  int t = blockIdx.x * blockDim.x + threadIdx.x;
  if (t >= R * D) return;
  int r = t / D;
  q32[t] = (r < NQ) ? qe[t] : 0.0f;
}

__global__ void k_layernorm128(const float* __restrict__ x,
                               const float* __restrict__ g,
                               const float* __restrict__ b,
                               h16* __restrict__ outh) {
  int r = blockIdx.x, lane = threadIdx.x;
  const float* xr = x + (size_t)r * 128;
  float v0 = xr[lane], v1 = xr[32 + lane], v2 = xr[64 + lane], v3 = xr[96 + lane];
  float s = v0 + v1 + v2 + v3;
  for (int o = 16; o; o >>= 1) s += __shfl_xor(s, o, 32);
  float mean = s * (1.0f / 128.0f);
  float d0 = v0 - mean, d1 = v1 - mean, d2 = v2 - mean, d3 = v3 - mean;
  float vv = d0 * d0 + d1 * d1 + d2 * d2 + d3 * d3;
  for (int o = 16; o; o >>= 1) vv += __shfl_xor(vv, o, 32);
  float rstd = rsqrtf(vv * (1.0f / 128.0f) + 1e-5f);
  h16* orow = outh + (size_t)r * 128;
  orow[lane]      = (h16)(d0 * rstd * g[lane] + b[lane]);
  orow[32 + lane] = (h16)(d1 * rstd * g[32 + lane] + b[32 + lane]);
  orow[64 + lane] = (h16)(d2 * rstd * g[64 + lane] + b[64 + lane]);
  orow[96 + lane] = (h16)(d3 * rstd * g[96 + lane] + b[96 + lane]);
}

__global__ void k_attn_scores(const float* __restrict__ Q, const float* __restrict__ Km,
                              float* __restrict__ att, int Lq, int Lk,
                              int H, int hd, int validK, float scale) {
  int t = blockIdx.x * blockDim.x + threadIdx.x;
  if (t >= H * Lq * Lk) return;
  int h = t / (Lq * Lk);
  int rem = t % (Lq * Lk);
  int i = rem / Lk, j = rem % Lk;
  float s;
  if (j >= validK) {
    s = -1e30f;
  } else {
    s = 0.0f;
    const float* qp = Q + (size_t)i * H * hd + h * hd;
    const float* kp = Km + (size_t)j * H * hd + h * hd;
    for (int d = 0; d < hd; ++d) s += qp[d] * kp[d];
    s *= scale;
  }
  att[t] = s;
}

__global__ void k_softmax(float* __restrict__ att, int Lk) {
  int row = blockIdx.x, lane = threadIdx.x;
  float* p = att + (size_t)row * Lk;
  float mx = -3.4e38f;
  for (int j = lane; j < Lk; j += 32) mx = fmaxf(mx, p[j]);
  for (int o = 16; o; o >>= 1) mx = fmaxf(mx, __shfl_xor(mx, o, 32));
  float s = 0.0f;
  for (int j = lane; j < Lk; j += 32) { float e = __expf(p[j] - mx); p[j] = e; s += e; }
  for (int o = 16; o; o >>= 1) s += __shfl_xor(s, o, 32);
  float inv = 1.0f / s;
  for (int j = lane; j < Lk; j += 32) p[j] *= inv;
}

__global__ void k_attn_out(const float* __restrict__ att, const float* __restrict__ V,
                           float* __restrict__ of, h16* __restrict__ oh,
                           int Lq, int Lk, int H, int hd) {
  int t = blockIdx.x * blockDim.x + threadIdx.x;
  int D = H * hd;
  if (t >= Lq * D) return;
  int i = t / D, c = t % D, h = c / hd;
  const float* a = att + ((size_t)h * Lq + i) * Lk;
  float s = 0.0f;
  for (int j = 0; j < Lk; ++j) s += a[j] * V[(size_t)j * D + c];
  of[t] = s;
  oh[t] = (h16)s;
}

__global__ void k_add(float* __restrict__ y, const float* __restrict__ x, int n) {
  int t = blockIdx.x * blockDim.x + threadIdx.x;
  if (t < n) y[t] += x[t];
}

__global__ void k_write_out(const float* __restrict__ q32,
                            const float* __restrict__ clsW, const float* __restrict__ clsb,
                            const float* __restrict__ masksF,
                            float* __restrict__ out, int NQ, int D, int Npts) {
  int t = blockIdx.x * blockDim.x + threadIdx.x;
  int total = NQ * 2 + NQ * Npts;
  if (t >= total) return;
  if (t < NQ * 2) {
    int q = t >> 1, o = t & 1;
    float s = clsb[o];
    const float* qp = q32 + (size_t)q * D;
    const float* wp = clsW + (size_t)o * D;
    for (int d = 0; d < D; ++d) s += qp[d] * wp[d];
    out[t] = s;
  } else {
    int r = t - NQ * 2;
    int q = r / Npts, j = r % Npts;
    out[t] = masksF[(size_t)j * 32 + q] * 0.08838834764831845f;  // 1/sqrt(128)
  }
}

// ===========================================================================
// Host side
// ===========================================================================
static inline int cdiv(int a, int b) { return (a + b - 1) / b; }
static inline void* carve(uint8_t*& p, size_t bytes) {
  void* r = (void*)p;
  p += (bytes + 255) & ~(size_t)255;
  return r;
}

extern "C" void kernel_launch(void* const* d_in, const int* in_sizes, int n_in,
                              void* d_out, int out_size, void* d_ws, size_t ws_size,
                              hipStream_t stream) {
  (void)in_sizes; (void)n_in; (void)out_size; (void)ws_size;
#define PF(i) ((const float*)d_in[(i)])
#define PI(i) ((const int*)d_in[(i)])

  // ----- problem dims -----
  const int N0 = 16384, S1 = 8192, S2 = 2048, S3 = 512, KNN = 32;
  const int D = 128, NQ = 20, QR = 32, H = 4, HD = 32, FFN = 512;
  const float INV_BN = 0.999995000037f;  // 1/sqrt(1+1e-5)

  // ----- input index map (jax tree-flatten: dict keys sorted, lists in order)
  // 0: features | 1: cls_W | 2: cls_b | 3..80: dec (3 x 26) |
  // 81..86: fp1 | 87..92: fp2 | 93..98: fp3 | 99: proj_W | 100: proj_b |
  // 101: query_embed | 102..110: sa1 | 111..119: sa2 | 120..128: sa3 |
  // 129..131: perm1..3
  const int IX_FEAT = 0, IX_CLSW = 1, IX_CLSB = 2, IX_DEC = 3;
  const int IX_FP1 = 81, IX_FP2 = 87, IX_FP3 = 93;
  const int IX_PROJW = 99, IX_PROJB = 100, IX_QE = 101;
  const int IX_SA1 = 102, IX_SA2 = 111, IX_SA3 = 120;
  const int IX_P1 = 129, IX_P2 = 130, IX_P3 = 131;
  // per-dec-layer offsets: W1=0 W2=1 b1=2 b2=3 | ca Wk..bv=4..11 |
  // ln1=12,13 ln2=14,15 ln3=16,17 | sa Wk..bv=18..25

  // ----- workspace carve -----
  uint8_t* wp = (uint8_t*)d_ws;
  h16* sa1W[3]; h16* sa2W[3]; h16* sa3W[3];
  const int sa1M[3] = {64, 64, 128},   sa1K[3] = {32, 64, 64};
  const int sa2M[3] = {128, 128, 256}, sa2K[3] = {160, 128, 128};
  const int sa3M[3] = {256, 256, 512}, sa3K[3] = {288, 256, 256};
  for (int i = 0; i < 3; ++i) sa1W[i] = (h16*)carve(wp, (size_t)sa1M[i] * sa1K[i] * 2);
  for (int i = 0; i < 3; ++i) sa2W[i] = (h16*)carve(wp, (size_t)sa2M[i] * sa2K[i] * 2);
  for (int i = 0; i < 3; ++i) sa3W[i] = (h16*)carve(wp, (size_t)sa3M[i] * sa3K[i] * 2);
  h16* fp3W[2]; h16* fp2W[2]; h16* fp1W[2];
  const int fp3M[2] = {256, 256}, fp3K[2] = {768, 256};
  const int fp2M[2] = {128, 128}, fp2K[2] = {384, 128};
  const int fp1M[2] = {128, 128}, fp1K[2] = {160, 128};
  for (int i = 0; i < 2; ++i) fp3W[i] = (h16*)carve(wp, (size_t)fp3M[i] * fp3K[i] * 2);
  for (int i = 0; i < 2; ++i) fp2W[i] = (h16*)carve(wp, (size_t)fp2M[i] * fp2K[i] * 2);
  for (int i = 0; i < 2; ++i) fp1W[i] = (h16*)carve(wp, (size_t)fp1M[i] * fp1K[i] * 2);
  h16* projP = (h16*)carve(wp, (size_t)128 * 512 * 2);
  h16* decW[3][10];
  for (int l = 0; l < 3; ++l) {
    for (int i = 0; i < 8; ++i) decW[l][i] = (h16*)carve(wp, (size_t)128 * 128 * 2);
    decW[l][8] = (h16*)carve(wp, (size_t)512 * 128 * 2);
    decW[l][9] = (h16*)carve(wp, (size_t)128 * 512 * 2);
  }
  h16* qryP = (h16*)carve(wp, (size_t)QR * 128 * 2);

  h16*  featrows0 = (h16*)carve(wp, (size_t)N0 * 16 * 2);
  float* xyz1 = (float*)carve(wp, (size_t)3 * S1 * 4);
  float* xyz2 = (float*)carve(wp, (size_t)3 * S2 * 4);
  float* xyz3 = (float*)carve(wp, (size_t)3 * S3 * 4);
  int*   idxb = (int*)carve(wp, (size_t)S1 * KNN * 4);
  h16*   XG = (h16*)carve(wp, (size_t)S2 * KNN * 160 * 2);
  h16*   YA = (h16*)carve(wp, (size_t)S1 * KNN * 128 * 2);
  h16*   YB = (h16*)carve(wp, (size_t)S1 * KNN * 128 * 2);
  h16*   f1 = (h16*)carve(wp, (size_t)S1 * 128 * 2);
  h16*   f2 = (h16*)carve(wp, (size_t)S2 * 256 * 2);
  h16*   f3 = (h16*)carve(wp, (size_t)S3 * 512 * 2);
  h16*   f2up = (h16*)carve(wp, (size_t)S2 * 256 * 2);
  h16*   f1up = (h16*)carve(wp, (size_t)S1 * 128 * 2);
  h16*   f0 = (h16*)carve(wp, (size_t)N0 * 128 * 2);
  h16*   mem_h = (h16*)carve(wp, (size_t)512 * 128 * 2);
  float* q32 = (float*)carve(wp, (size_t)QR * 128 * 4);
  h16*   qn_h = (h16*)carve(wp, (size_t)QR * 128 * 2);
  float* Qf = (float*)carve(wp, (size_t)QR * 128 * 4);
  float* Kf = (float*)carve(wp, (size_t)512 * 128 * 4);
  float* Vf = (float*)carve(wp, (size_t)512 * 128 * 4);
  float* att = (float*)carve(wp, (size_t)H * QR * 512 * 4);
  float* attO_f = (float*)carve(wp, (size_t)QR * 128 * 4);
  h16*   attO_h = (h16*)carve(wp, (size_t)QR * 128 * 2);
  h16*   ffn_h = (h16*)carve(wp, (size_t)QR * FFN * 2);
  float* projF = (float*)carve(wp, (size_t)QR * 128 * 4);
  float* masksF = (float*)carve(wp, (size_t)N0 * QR * 4);

  // ----- helpers -----
  auto prepw = [&](h16* dst, int widx, int gidx, int M, int K, int Mp, int Kp) {
    const float* g = (gidx >= 0) ? PF(gidx) : nullptr;
    int n = Mp * Kp;
    k_prep_weight<<<cdiv(n, 256), 256, 0, stream>>>(PF(widx), g, dst, M, K, Mp, Kp, INV_BN);
  };
  auto gemm = [&](const h16* W, const h16* X, const float* b, h16* Yh, float* Yf,
                  int Mp, int Kp, int L, int relu) {
    if (L % 64 == 0)
      k_gemm<4><<<dim3(L / 64, Mp / 16), dim3(32), 0, stream>>>(W, X, b, Yh, Yf, Mp, Kp, relu);
    else
      k_gemm<2><<<dim3(L / 32, Mp / 16), dim3(32), 0, stream>>>(W, X, b, Yh, Yf, Mp, Kp, relu);
  };

  // ----- weight prep -----
  const int sa1src[3] = {IX_SA1, IX_SA1 + 3, IX_SA1 + 6};
  const int sa2src[3] = {IX_SA2, IX_SA2 + 3, IX_SA2 + 6};
  const int sa3src[3] = {IX_SA3, IX_SA3 + 3, IX_SA3 + 6};
  const int sa1kc[3] = {12, 64, 64}, sa2kc[3] = {131, 128, 128}, sa3kc[3] = {259, 256, 256};
  for (int i = 0; i < 3; ++i) {
    prepw(sa1W[i], sa1src[i], sa1src[i] + 1, sa1M[i], sa1kc[i], sa1M[i], sa1K[i]);
    prepw(sa2W[i], sa2src[i], sa2src[i] + 1, sa2M[i], sa2kc[i], sa2M[i], sa2K[i]);
    prepw(sa3W[i], sa3src[i], sa3src[i] + 1, sa3M[i], sa3kc[i], sa3M[i], sa3K[i]);
  }
  const int fp3kc[2] = {768, 256}, fp2kc[2] = {384, 128}, fp1kc[2] = {137, 128};
  for (int i = 0; i < 2; ++i) {
    prepw(fp3W[i], IX_FP3 + 3 * i, IX_FP3 + 3 * i + 1, fp3M[i], fp3kc[i], fp3M[i], fp3K[i]);
    prepw(fp2W[i], IX_FP2 + 3 * i, IX_FP2 + 3 * i + 1, fp2M[i], fp2kc[i], fp2M[i], fp2K[i]);
    prepw(fp1W[i], IX_FP1 + 3 * i, IX_FP1 + 3 * i + 1, fp1M[i], fp1kc[i], fp1M[i], fp1K[i]);
  }
  prepw(projP, IX_PROJW, -1, 128, 512, 128, 512);
  for (int l = 0; l < 3; ++l) {
    int LB = IX_DEC + 26 * l;
    prepw(decW[l][0], LB + 20, -1, 128, 128, 128, 128);  // saWq
    prepw(decW[l][1], LB + 18, -1, 128, 128, 128, 128);  // saWk
    prepw(decW[l][2], LB + 21, -1, 128, 128, 128, 128);  // saWv
    prepw(decW[l][3], LB + 19, -1, 128, 128, 128, 128);  // saWo
    prepw(decW[l][4], LB + 6, -1, 128, 128, 128, 128);   // caWq
    prepw(decW[l][5], LB + 4, -1, 128, 128, 128, 128);   // caWk
    prepw(decW[l][6], LB + 7, -1, 128, 128, 128, 128);   // caWv
    prepw(decW[l][7], LB + 5, -1, 128, 128, 128, 128);   // caWo
    prepw(decW[l][8], LB + 0, -1, 512, 128, 512, 128);   // W1
    prepw(decW[l][9], LB + 1, -1, 128, 512, 128, 512);   // W2
  }

  // ----- backbone -----
  const float* feats = PF(IX_FEAT);  // (9, 16384); rows 0..2 are xyz
  k_cm_to_rows<<<cdiv(N0 * 16, 256), 256, 0, stream>>>(feats, 9, N0, featrows0, 16);
  k_gather_xyz<<<cdiv(3 * S1, 256), 256, 0, stream>>>(feats, N0, PI(IX_P1), xyz1, S1);

  // SA1
  k_knn<32><<<cdiv(S1, 64), 64, 0, stream>>>(feats, N0, xyz1, S1, idxb);
  {
    int L = S1 * KNN;
    k_group<<<cdiv(L, 256), 256, 0, stream>>>(featrows0, 16, 9, feats, N0, xyz1, S1,
                                              idxb, KNN, XG, 32);
    gemm(sa1W[0], XG, PF(IX_SA1 + 2), YA, nullptr, 64, 32, L, 1);
    gemm(sa1W[1], YA, PF(IX_SA1 + 5), YB, nullptr, 64, 64, L, 1);
    gemm(sa1W[2], YB, PF(IX_SA1 + 8), YA, nullptr, 128, 64, L, 1);
    k_maxk<<<cdiv(S1 * 128, 256), 256, 0, stream>>>(YA, f1, S1, KNN, 128);
  }
  // SA2
  k_gather_xyz<<<cdiv(3 * S2, 256), 256, 0, stream>>>(xyz1, S1, PI(IX_P2), xyz2, S2);
  k_knn<32><<<cdiv(S2, 64), 64, 0, stream>>>(xyz1, S1, xyz2, S2, idxb);
  {
    int L = S2 * KNN;
    k_group<<<cdiv(L, 256), 256, 0, stream>>>(f1, 128, 128, xyz1, S1, xyz2, S2,
                                              idxb, KNN, XG, 160);
    gemm(sa2W[0], XG, PF(IX_SA2 + 2), YA, nullptr, 128, 160, L, 1);
    gemm(sa2W[1], YA, PF(IX_SA2 + 5), YB, nullptr, 128, 128, L, 1);
    gemm(sa2W[2], YB, PF(IX_SA2 + 8), YA, nullptr, 256, 128, L, 1);
    k_maxk<<<cdiv(S2 * 256, 256), 256, 0, stream>>>(YA, f2, S2, KNN, 256);
  }
  // SA3
  k_gather_xyz<<<cdiv(3 * S3, 256), 256, 0, stream>>>(xyz2, S2, PI(IX_P3), xyz3, S3);
  k_knn<32><<<cdiv(S3, 64), 64, 0, stream>>>(xyz2, S2, xyz3, S3, idxb);
  {
    int L = S3 * KNN;
    k_group<<<cdiv(L, 256), 256, 0, stream>>>(f2, 256, 256, xyz2, S2, xyz3, S3,
                                              idxb, KNN, XG, 288);
    gemm(sa3W[0], XG, PF(IX_SA3 + 2), YA, nullptr, 256, 288, L, 1);
    gemm(sa3W[1], YA, PF(IX_SA3 + 5), YB, nullptr, 256, 256, L, 1);
    gemm(sa3W[2], YB, PF(IX_SA3 + 8), YA, nullptr, 512, 256, L, 1);
    k_maxk<<<cdiv(S3 * 512, 256), 256, 0, stream>>>(YA, f3, S3, KNN, 512);
  }

  // FP3
  k_knn<3><<<cdiv(S2, 64), 64, 0, stream>>>(xyz3, S3, xyz2, S2, idxb);
  k_interp_concat<<<cdiv(S2, 128), 128, 0, stream>>>(xyz2, S2, xyz3, S3, idxb, 3,
                                                     f3, 512, 512, f2, 256, 256, XG, 768);
  gemm(fp3W[0], XG, PF(IX_FP3 + 2), YA, nullptr, 256, 768, S2, 1);
  gemm(fp3W[1], YA, PF(IX_FP3 + 5), f2up, nullptr, 256, 256, S2, 1);
  // FP2
  k_knn<3><<<cdiv(S1, 64), 64, 0, stream>>>(xyz2, S2, xyz1, S1, idxb);
  k_interp_concat<<<cdiv(S1, 128), 128, 0, stream>>>(xyz1, S1, xyz2, S2, idxb, 3,
                                                     f2up, 256, 256, f1, 128, 128, XG, 384);
  gemm(fp2W[0], XG, PF(IX_FP2 + 2), YA, nullptr, 128, 384, S1, 1);
  gemm(fp2W[1], YA, PF(IX_FP2 + 5), f1up, nullptr, 128, 128, S1, 1);
  // FP1
  k_knn<3><<<cdiv(N0, 64), 64, 0, stream>>>(xyz1, S1, feats, N0, idxb);
  k_interp_concat<<<cdiv(N0, 128), 128, 0, stream>>>(feats, N0, xyz1, S1, idxb, 3,
                                                     f1up, 128, 128, featrows0, 16, 9, XG, 160);
  gemm(fp1W[0], XG, PF(IX_FP1 + 2), YA, nullptr, 128, 160, N0, 1);
  gemm(fp1W[1], YA, PF(IX_FP1 + 5), f0, nullptr, 128, 128, N0, 1);

  // memory = f3 @ proj_W.T + proj_b -> [512, 128] f16 rows
  gemm(projP, f3, PF(IX_PROJB), mem_h, nullptr, 128, 512, 512, 0);

  // ----- decoder -----
  k_init_q<<<cdiv(QR * D, 256), 256, 0, stream>>>(PF(IX_QE), q32, NQ, D, QR);
  const float scaleH = 0.1767766952966369f;  // 1/sqrt(32)
  for (int l = 0; l < 3; ++l) {
    int LB = IX_DEC + 26 * l;
    // --- self attention ---
    k_layernorm128<<<QR, 32, 0, stream>>>(q32, PF(LB + 12), PF(LB + 13), qn_h);
    gemm(decW[l][0], qn_h, PF(LB + 24), nullptr, Qf, 128, 128, QR, 0);
    gemm(decW[l][1], qn_h, PF(LB + 22), nullptr, Kf, 128, 128, QR, 0);
    gemm(decW[l][2], qn_h, PF(LB + 25), nullptr, Vf, 128, 128, QR, 0);
    k_attn_scores<<<cdiv(H * QR * QR, 256), 256, 0, stream>>>(Qf, Kf, att, QR, QR, H, HD,
                                                              NQ, scaleH);
    k_softmax<<<H * QR, 32, 0, stream>>>(att, QR);
    k_attn_out<<<cdiv(QR * D, 256), 256, 0, stream>>>(att, Vf, attO_f, attO_h, QR, QR, H, HD);
    gemm(decW[l][3], attO_h, PF(LB + 23), nullptr, projF, 128, 128, QR, 0);
    k_add<<<cdiv(QR * D, 256), 256, 0, stream>>>(q32, projF, QR * D);
    // --- cross attention ---
    k_layernorm128<<<QR, 32, 0, stream>>>(q32, PF(LB + 14), PF(LB + 15), qn_h);
    gemm(decW[l][4], qn_h, PF(LB + 10), nullptr, Qf, 128, 128, QR, 0);
    gemm(decW[l][5], mem_h, PF(LB + 8), nullptr, Kf, 128, 128, 512, 0);
    gemm(decW[l][6], mem_h, PF(LB + 11), nullptr, Vf, 128, 128, 512, 0);
    k_attn_scores<<<cdiv(H * QR * 512, 256), 256, 0, stream>>>(Qf, Kf, att, QR, 512, H, HD,
                                                               512, scaleH);
    k_softmax<<<H * QR, 32, 0, stream>>>(att, 512);
    k_attn_out<<<cdiv(QR * D, 256), 256, 0, stream>>>(att, Vf, attO_f, attO_h, QR, 512, H, HD);
    gemm(decW[l][7], attO_h, PF(LB + 9), nullptr, projF, 128, 128, QR, 0);
    k_add<<<cdiv(QR * D, 256), 256, 0, stream>>>(q32, projF, QR * D);
    // --- FFN ---
    k_layernorm128<<<QR, 32, 0, stream>>>(q32, PF(LB + 16), PF(LB + 17), qn_h);
    gemm(decW[l][8], qn_h, PF(LB + 2), ffn_h, nullptr, 512, 128, QR, 1);
    gemm(decW[l][9], ffn_h, PF(LB + 3), nullptr, projF, 128, 512, QR, 0);
    k_add<<<cdiv(QR * D, 256), 256, 0, stream>>>(q32, projF, QR * D);
  }

  // ----- heads -----
  k_prep_weight<<<cdiv(QR * 128, 256), 256, 0, stream>>>(q32, nullptr, qryP,
                                                         QR, 128, QR, 128, 1.0f);
  gemm(qryP, f0, nullptr, nullptr, masksF, QR, 128, N0, 0);
  int total = NQ * 2 + NQ * N0;
  k_write_out<<<cdiv(total, 256), 256, 0, stream>>>(q32, PF(IX_CLSW), PF(IX_CLSB),
                                                    masksF, (float*)d_out, NQ, D, N0);
#undef PF
#undef PI
}
